// TextSAGEDask_51565377356535
// MI455X (gfx1250) — matrix operand order repaired and checked by reading, not compiled, not run on hardware
//
#include <hip/hip_runtime.h>

typedef __attribute__((ext_vector_type(16))) __bf16 v16bf;
typedef __attribute__((ext_vector_type(8)))  __bf16 v8bf;
typedef __attribute__((ext_vector_type(8)))  float  v8f;

#define LDK  136   // transposed-B per-column k stride (K=128 + 8 pad), bf16 elems
#define LDKC 40    // transposed-B per-column k stride for one 32-K chunk (32 + 8 pad)
#define CHUNK_BYTES (128 * LDKC * 2)   // one packed projW chunk image (10240 B)
#define FULL_BYTES  (128 * LDK * 2)    // one packed 128-K weight image (34816 B)

__device__ __forceinline__ v8f wmma_bf16(v16bf a, v16bf b, v8f c) {
  return __builtin_amdgcn_wmma_f32_16x16x32_bf16(
      /*neg_a=*/false, a, /*neg_b=*/false, b,
      /*c_mod=*/(short)0, c, /*reuse_a=*/false, /*reuse_b=*/false);
}

// ---- CDNA5 async global->LDS staging (per-lane 16B, tracked by ASYNCcnt) ----
__device__ __forceinline__ void async_b128(unsigned ldsAddr, unsigned long long gaddr) {
  asm volatile("global_load_async_to_lds_b128 %0, %1, off"
               :: "v"(ldsAddr), "v"(gaddr)
               : "memory");
}

__device__ __forceinline__ void wait_async0() {
  asm volatile("s_wait_asynccnt 0" ::: "memory");
}

// Block-wide linear copy of `bytes` (multiple of 16) from global to LDS.
__device__ __forceinline__ void stage_async(const __bf16* __restrict__ gsrc,
                                            __bf16* lds, int bytes) {
  const unsigned base = (unsigned)(uintptr_t)(void*)lds;        // LDS byte offset
  const unsigned long long g = (unsigned long long)(uintptr_t)gsrc;
  for (int off = (int)threadIdx.x * 16; off < bytes; off += 256 * 16)
    async_b128(base + off, g + off);
}

// B fragment from transposed LDS (BsT[col][k]): lane holds column n=lane&15,
// K = k0 + {0..7} and k0 + 16 + {0..7} -> two ds_load_b128.
__device__ __forceinline__ v16bf bfragT(const __bf16* BsT, int ldk, int col, int k0) {
  const v8bf* p = (const v8bf*)(BsT + col * ldk + k0);
  v8bf lo = p[0];
  v8bf hi = p[2];
  return __builtin_shufflevector(lo, hi, 0, 1, 2, 3, 4, 5, 6, 7,
                                 8, 9, 10, 11, 12, 13, 14, 15);
}

// A fragment from a row-major f32 row pointer: lane holds row m=lane&15,
// K = k0 + kbA + {0..7} and k0 + 16 + kbA + {0..7}, kbA = 8*(lane>>4).
__device__ __forceinline__ v16bf afrag_row(const float* rowp, int k0, int kbA, float scale) {
  float4 u0 = *(const float4*)(rowp + k0 + kbA);
  float4 u1 = *(const float4*)(rowp + k0 + kbA + 4);
  float4 u2 = *(const float4*)(rowp + k0 + 16 + kbA);
  float4 u3 = *(const float4*)(rowp + k0 + 16 + kbA + 4);
  v16bf a;
  a[0]  = (__bf16)(u0.x * scale); a[1]  = (__bf16)(u0.y * scale);
  a[2]  = (__bf16)(u0.z * scale); a[3]  = (__bf16)(u0.w * scale);
  a[4]  = (__bf16)(u1.x * scale); a[5]  = (__bf16)(u1.y * scale);
  a[6]  = (__bf16)(u1.z * scale); a[7]  = (__bf16)(u1.w * scale);
  a[8]  = (__bf16)(u2.x * scale); a[9]  = (__bf16)(u2.y * scale);
  a[10] = (__bf16)(u2.z * scale); a[11] = (__bf16)(u2.w * scale);
  a[12] = (__bf16)(u3.x * scale); a[13] = (__bf16)(u3.y * scale);
  a[14] = (__bf16)(u3.z * scale); a[15] = (__bf16)(u3.w * scale);
  return a;
}

// ---------------- small utility kernels ----------------

__global__ void k_zero(float* __restrict__ p, int count) {
  int i = blockIdx.x * blockDim.x + threadIdx.x;
  if (i < count) p[i] = 0.0f;
}

__global__ void k_deg(const int* __restrict__ dst, int E, float* __restrict__ deg) {
  int e = blockIdx.x * blockDim.x + threadIdx.x;
  if (e < E) atomicAdd(&deg[dst[e]], 1.0f);
}

// One-shot weight repack: W f32 [nchunks*chunkK, 128] row-major ->
// transposed bf16 chunk images out[c][col][k] with per-column stride ldk
// (exact LDS layout, so GEMM staging becomes a linear memcpy).
__global__ void k_pack(const float* __restrict__ W, __bf16* __restrict__ out,
                       int chunkK, int ldk, int nchunks) {
  int idx = blockIdx.x * blockDim.x + threadIdx.x;
  int total = nchunks * 128 * chunkK;
  if (idx >= total) return;
  int k   = idx % chunkK;
  int col = (idx / chunkK) & 127;
  int c   = idx / (chunkK * 128);
  out[(c * 128 + col) * ldk + k] = (__bf16)W[(c * chunkK + k) * 128 + col];
}

// Edge scatter: thread = (edge, 4-dim chunk). float4 gather from x[src],
// 4x global_atomic_add_f32 into agg[dst]. All traffic is L2-resident.
__global__ void k_scatter(const int* __restrict__ src, const int* __restrict__ dst, int E,
                          const float* __restrict__ x, float* __restrict__ agg) {
  int tid = blockIdx.x * blockDim.x + threadIdx.x;
  if (tid >= E * 32) return;
  int e = tid >> 5;
  int c = (tid & 31) << 2;
  int s = src[e], d = dst[e];
  float4 v = *(const float4*)(x + (long long)s * 128 + c);
  float* ap = agg + (long long)d * 128 + c;
  atomicAdd(ap + 0, v.x);
  atomicAdd(ap + 1, v.y);
  atomicAdd(ap + 2, v.z);
  atomicAdd(ap + 3, v.w);
}

// ---------------- fused gather + projection GEMM ----------------
// x[row] = concat([mean(featEmb[ids]) (128), mean-word text (3*64)]) @ projW(320x128)
//          + projb + idEmb[row]
// 8 waves/block, wave owns a 16-row tile. K=320 in 10 chunks of 32; packed projW
// chunks are double-buffered into LDS with async-to-LDS copies overlapped with
// the gather-mean A-fragment construction.
__global__ __launch_bounds__(256) void k_node_init(
    const int* __restrict__ featIds, const float* __restrict__ featEmb,
    const int* __restrict__ textIds, const float* __restrict__ wordEmb,
    const __bf16* __restrict__ packedProj, const float* __restrict__ projb,
    const float* __restrict__ idEmb, float* __restrict__ xOut, int nrows) {
  __shared__ __bf16 Bc[2][128 * LDKC];  // 2 x 10 KB, transposed: Bc[buf][col][k]

  const int wave = threadIdx.x >> 5;
  const int lane = threadIdx.x & 31;
  const int m    = lane & 15;
  const int half = lane >> 4;
  const int kbA  = half * 8;
  const int kbB  = half * 8;
  const int rowtile = blockIdx.x * 128 + wave * 16;
  int row = rowtile + m;
  if (row >= nrows) row = nrows - 1;   // clamp; stores are guarded

  int fid[10];
#pragma unroll
  for (int j = 0; j < 10; ++j) fid[j] = featIds[(long long)row * 10 + j];
  int tw[24];
#pragma unroll
  for (int j = 0; j < 24; ++j) tw[j] = textIds[(long long)row * 24 + j];

  v8f acc[8] = {};

  stage_async(packedProj, Bc[0], CHUNK_BYTES);   // prefetch chunk 0

  for (int c = 0; c < 10; ++c) {
    wait_async0();
    __syncthreads();   // buffer c&1 fully staged; all waves done with buffer (c+1)&1
    if (c < 9)
      stage_async(packedProj + (c + 1) * 128 * LDKC, Bc[(c + 1) & 1], CHUNK_BYTES);
    const __bf16* Bcur = Bc[c & 1];

    // Build this lane's A fragment for chunk c directly from the gathers.
    float v[16];
#pragma unroll
    for (int i = 0; i < 16; ++i) v[i] = 0.0f;
    float scale;
    if (c < 4) {  // feature-embedding mean, K in [0,128)
      const int kc = c * 32;
#pragma unroll
      for (int j = 0; j < 10; ++j) {
        const float* fr = featEmb + (long long)fid[j] * 128 + kc;
        float4 p0 = *(const float4*)(fr + kbA);
        float4 p1 = *(const float4*)(fr + kbA + 4);
        float4 p2 = *(const float4*)(fr + 16 + kbA);
        float4 p3 = *(const float4*)(fr + 16 + kbA + 4);
        v[0] += p0.x;  v[1] += p0.y;  v[2] += p0.z;  v[3] += p0.w;
        v[4] += p1.x;  v[5] += p1.y;  v[6] += p1.z;  v[7] += p1.w;
        v[8] += p2.x;  v[9] += p2.y;  v[10] += p2.z; v[11] += p2.w;
        v[12] += p3.x; v[13] += p3.y; v[14] += p3.z; v[15] += p3.w;
      }
      scale = 0.1f;
    } else {      // text mean, K in [128,320): field f, dim within 64
      const int f = (c - 4) >> 1;
      const int dbase = ((c - 4) & 1) * 32;
#pragma unroll
      for (int l = 0; l < 8; ++l) {
        const float* wr = wordEmb + (long long)tw[f * 8 + l] * 64 + dbase;
        float4 p0 = *(const float4*)(wr + kbA);
        float4 p1 = *(const float4*)(wr + kbA + 4);
        float4 p2 = *(const float4*)(wr + 16 + kbA);
        float4 p3 = *(const float4*)(wr + 16 + kbA + 4);
        v[0] += p0.x;  v[1] += p0.y;  v[2] += p0.z;  v[3] += p0.w;
        v[4] += p1.x;  v[5] += p1.y;  v[6] += p1.z;  v[7] += p1.w;
        v[8] += p2.x;  v[9] += p2.y;  v[10] += p2.z; v[11] += p2.w;
        v[12] += p3.x; v[13] += p3.y; v[14] += p3.z; v[15] += p3.w;
      }
      scale = 0.125f;
    }
    v16bf a;
#pragma unroll
    for (int i = 0; i < 16; ++i) a[i] = (__bf16)(v[i] * scale);

#pragma unroll
    for (int t = 0; t < 8; ++t) {
      v16bf b = bfragT(Bcur, LDKC, t * 16 + m, kbB);
      acc[t] = wmma_bf16(a, b, acc[t]);
    }
  }

  // Epilogue: D[j] -> row = rowtile + j + 8*half, col = t*16 + (lane&15)
#pragma unroll
  for (int t = 0; t < 8; ++t) {
    const int col = t * 16 + m;
    const float bb = projb[col];
#pragma unroll
    for (int j = 0; j < 8; ++j) {
      int r = rowtile + j + 8 * half;
      if (r < nrows)
        xOut[(long long)r * 128 + col] = acc[t][j] + bb + idEmb[(long long)r * 128 + col];
    }
  }
}

// ---------------- h = relu((agg/deg) @ vW + vb), in place on agg ----------------
__global__ __launch_bounds__(256) void k_h(
    float* __restrict__ agg, const float* __restrict__ deg,
    const __bf16* __restrict__ packedV, const float* __restrict__ vb, int Nn) {
  __shared__ __bf16 Bs[128 * LDK];  // ~34 KB, transposed: Bs[col][k]
  stage_async(packedV, Bs, FULL_BYTES);

  const int wave = threadIdx.x >> 5;
  const int lane = threadIdx.x & 31;
  const int m    = lane & 15;
  const int half = lane >> 4;
  const int kbA  = half * 8;
  const int kbB  = half * 8;
  const int rowtile = blockIdx.x * 128 + wave * 16;
  int row = rowtile + m;
  if (row >= Nn) row = Nn - 1;

  const float invd = 1.0f / fmaxf(deg[row], 1.0f);
  const float* rowp = agg + (long long)row * 128;
  v16bf aF[4];
#pragma unroll
  for (int c = 0; c < 4; ++c) aF[c] = afrag_row(rowp, c * 32, kbA, invd);

  wait_async0();
  __syncthreads();

  v8f acc[8] = {};
#pragma unroll
  for (int c = 0; c < 4; ++c)
#pragma unroll
    for (int t = 0; t < 8; ++t)
      acc[t] = wmma_bf16(aF[c], bfragT(Bs, LDK, t * 16 + m, c * 32 + kbB), acc[t]);

#pragma unroll
  for (int t = 0; t < 8; ++t) {
    const int col = t * 16 + m;
    const float bb = vb[col];
#pragma unroll
    for (int j = 0; j < 8; ++j) {
      int r = rowtile + j + 8 * half;
      if (r < Nn) agg[(long long)r * 128 + col] = fmaxf(acc[t][j] + bb, 0.0f);
    }
  }
}

// ---------------- x = concat([x, h]) @ wW + wb (+relu), in place on x ----------------
__global__ __launch_bounds__(256) void k_out(
    float* __restrict__ x, const float* __restrict__ h,
    const __bf16* __restrict__ packedW, const float* __restrict__ wb, int Nn, int doRelu) {
  __shared__ __bf16 Bs[128 * LDK];  // transposed: Bs[col][k], reused for both halves

  const int wave = threadIdx.x >> 5;
  const int lane = threadIdx.x & 31;
  const int m    = lane & 15;
  const int half = lane >> 4;
  const int kbA  = half * 8;
  const int kbB  = half * 8;
  const int rowtile = blockIdx.x * 128 + wave * 16;
  int row = rowtile + m;
  if (row >= Nn) row = Nn - 1;

  const float* xr = x + (long long)row * 128;
  const float* hr = h + (long long)row * 128;
  v8f acc[8] = {};

  // Phase 1: top half of wW (multiplies x)
  stage_async(packedW, Bs, FULL_BYTES);
  {
    v16bf aF[4];
#pragma unroll
    for (int c = 0; c < 4; ++c) aF[c] = afrag_row(xr, c * 32, kbA, 1.0f);
    wait_async0();
    __syncthreads();
#pragma unroll
    for (int c = 0; c < 4; ++c)
#pragma unroll
      for (int t = 0; t < 8; ++t)
        acc[t] = wmma_bf16(aF[c], bfragT(Bs, LDK, t * 16 + m, c * 32 + kbB), acc[t]);
  }
  __syncthreads();

  // Phase 2: bottom half of wW (multiplies h)
  stage_async(packedW + 128 * LDK, Bs, FULL_BYTES);
  {
    v16bf aF[4];
#pragma unroll
    for (int c = 0; c < 4; ++c) aF[c] = afrag_row(hr, c * 32, kbA, 1.0f);
    wait_async0();
    __syncthreads();
#pragma unroll
    for (int c = 0; c < 4; ++c)
#pragma unroll
      for (int t = 0; t < 8; ++t)
        acc[t] = wmma_bf16(aF[c], bfragT(Bs, LDK, t * 16 + m, c * 32 + kbB), acc[t]);
  }

#pragma unroll
  for (int t = 0; t < 8; ++t) {
    const int col = t * 16 + m;
    const float bb = wb[col];
#pragma unroll
    for (int j = 0; j < 8; ++j) {
      int r = rowtile + j + 8 * half;
      if (r < Nn) {
        float val = acc[t][j] + bb;
        if (doRelu) val = fmaxf(val, 0.0f);
        x[(long long)r * 128 + col] = val;
      }
    }
  }
}

extern "C" void kernel_launch(void* const* d_in, const int* in_sizes, int n_in,
                              void* d_out, int out_size, void* d_ws, size_t ws_size,
                              hipStream_t stream) {
  const int E_  = in_sizes[0] / 2;
  const int NU_ = in_sizes[1] / 10;
  const int NI_ = in_sizes[2] / 10;
  const int Nn  = NU_ + NI_;

  const int*   edge  = (const int*)d_in[0];
  const int*   src   = edge;
  const int*   dst   = edge + E_;
  const int*   uFeat = (const int*)d_in[1];
  const int*   iFeat = (const int*)d_in[2];
  const int*   uText = (const int*)d_in[3];
  const int*   iText = (const int*)d_in[4];
  const float* uId   = (const float*)d_in[5];
  const float* iId   = (const float*)d_in[6];
  const float* uFE   = (const float*)d_in[7];
  const float* iFE   = (const float*)d_in[8];
  const float* wEmb  = (const float*)d_in[9];
  const float* uPW   = (const float*)d_in[10];
  const float* uPb   = (const float*)d_in[11];
  const float* iPW   = (const float*)d_in[12];
  const float* iPb   = (const float*)d_in[13];
  const float* w0W   = (const float*)d_in[14];
  const float* w0b   = (const float*)d_in[15];
  const float* w1W   = (const float*)d_in[16];
  const float* w1b   = (const float*)d_in[17];
  const float* v0W   = (const float*)d_in[18];
  const float* v0b   = (const float*)d_in[19];
  const float* v1W   = (const float*)d_in[20];
  const float* v1b   = (const float*)d_in[21];

  float* x   = (float*)d_out;                 // [N,128] f32, also final output
  float* agg = (float*)d_ws;                  // [N,128] f32
  float* deg = agg + (size_t)Nn * 128;        // [N] f32

  // packed transposed bf16 weight images (exact LDS layout), after deg
  __bf16* pack = (__bf16*)(deg + Nn);         // 16B-aligned (Nn*516 % 16 == 0)
  __bf16* ppU = pack;                         // 10 chunks * 128 * 40
  __bf16* ppI = ppU + 10 * 128 * LDKC;
  __bf16* pv0 = ppI + 10 * 128 * LDKC;        // 128 * 136 each
  __bf16* pv1 = pv0 + 128 * LDK;
  __bf16* pw0 = pv1 + 128 * LDK;              // 2 * 128 * 136 each
  __bf16* pw1 = pw0 + 2 * 128 * LDK;

  const int zc_all = Nn * 128 + Nn;
  const int zc_agg = Nn * 128;
  const int nodeBlocksU = (NU_ + 127) / 128;
  const int nodeBlocksI = (NI_ + 127) / 128;
  const int nodeBlocksN = (Nn + 127) / 128;
  const int scatBlocks  = (int)(((long long)E_ * 32 + 255) / 256);
  const int packProjT = 10 * 128 * 32, packVT = 128 * 128, packWT = 2 * 128 * 128;

  // one-shot weight repacks + zero + degree
  k_pack<<<(packProjT + 255) / 256, 256, 0, stream>>>(uPW, ppU, 32, LDKC, 10);
  k_pack<<<(packProjT + 255) / 256, 256, 0, stream>>>(iPW, ppI, 32, LDKC, 10);
  k_pack<<<(packVT + 255) / 256, 256, 0, stream>>>(v0W, pv0, 128, LDK, 1);
  k_pack<<<(packVT + 255) / 256, 256, 0, stream>>>(v1W, pv1, 128, LDK, 1);
  k_pack<<<(packWT + 255) / 256, 256, 0, stream>>>(w0W, pw0, 128, LDK, 2);
  k_pack<<<(packWT + 255) / 256, 256, 0, stream>>>(w1W, pw1, 128, LDK, 2);
  k_zero<<<(zc_all + 255) / 256, 256, 0, stream>>>(agg, zc_all);
  k_deg<<<(E_ + 255) / 256, 256, 0, stream>>>(dst, E_, deg);

  // initial node embeddings
  k_node_init<<<nodeBlocksU, 256, 0, stream>>>(uFeat, uFE, uText, wEmb, ppU, uPb, uId, x, NU_);
  k_node_init<<<nodeBlocksI, 256, 0, stream>>>(iFeat, iFE, iText, wEmb, ppI, iPb, iId,
                                               x + (size_t)NU_ * 128, NI_);
  // layer 0
  k_scatter<<<scatBlocks, 256, 0, stream>>>(src, dst, E_, x, agg);
  k_h<<<nodeBlocksN, 256, 0, stream>>>(agg, deg, pv0, v0b, Nn);
  k_out<<<nodeBlocksN, 256, 0, stream>>>(x, agg, pw0, w0b, Nn, 1);
  // layer 1
  k_zero<<<(zc_agg + 255) / 256, 256, 0, stream>>>(agg, zc_agg);
  k_scatter<<<scatBlocks, 256, 0, stream>>>(src, dst, E_, x, agg);
  k_h<<<nodeBlocksN, 256, 0, stream>>>(agg, deg, pv1, v1b, Nn);
  k_out<<<nodeBlocksN, 256, 0, stream>>>(x, agg, pw1, w1b, Nn, 0);
}
